// AGC_85126251807219
// MI455X (gfx1250) — matrix operand order repaired and checked by reading, not compiled, not run on hardware
//
#include <hip/hip_runtime.h>

typedef __bf16 bf16;
typedef unsigned short u16;
typedef __attribute__((ext_vector_type(16))) __bf16 v16bf;
typedef __attribute__((ext_vector_type(8)))  __bf16 v8bf;
typedef __attribute__((ext_vector_type(8)))  float  v8f;

#define N_ROWS 346112   // 512 * 676
#define NPB    676
#define BSZ    512
#define FDIM   128

// ---------------------------------------------------------------------------
// init: zero BN stat accumulators, set softmax scalars, convert weights->bf16
// wbf layout: W1[0..16383] W2[16384..32767] W3[32768..40959] W4[40960..45055]
// stats: 5 stages x 512 floats (sum[128] | sumsq[128] | scale[128] | shift[128])
// ---------------------------------------------------------------------------
__global__ __launch_bounds__(256) void k_init(const float* __restrict__ W1,
                                              const float* __restrict__ W2,
                                              const float* __restrict__ W3,
                                              const float* __restrict__ W4,
                                              bf16* __restrict__ wbf,
                                              float* __restrict__ stats,
                                              float* __restrict__ scal) {
  int t = blockIdx.x * 256 + threadIdx.x;
  if (t < 45056) {
    float v;
    if (t < 16384)      v = W1[t];
    else if (t < 32768) v = W2[t - 16384];
    else if (t < 40960) v = W3[t - 32768];
    else                v = W4[t - 40960];
    wbf[t] = (bf16)v;
  }
  if (t < 2560) stats[t] = 0.f;
  if (t == 0) { scal[0] = -__builtin_inff(); scal[1] = 0.f; }
}

// ---------------------------------------------------------------------------
// diff = |hub - x| in bf16 (vectorized x4).
// abs taken AFTER the f32->bf16 convert, in the integer domain (clear sign
// bit). RNE rounding commutes with abs, so this is bit-exact, and it removes
// every float-fabs pattern from the IR (the float fabs-of-sub + bf16-pack
// combination crashes gfx1250 ISel in this toolchain).
// ---------------------------------------------------------------------------
static __device__ __forceinline__ u16 bfabs_bits(float d) {
  bf16 h = (bf16)d;
  u16 u = __builtin_bit_cast(u16, h);
  return (u16)(u & 0x7FFFu);
}

__global__ __launch_bounds__(256) void k_diff(const float* __restrict__ x,
                                              u16* __restrict__ A0) {
  long idx = (long)blockIdx.x * 256 + threadIdx.x;
  long e = idx * 4;
  int r = (int)(e >> 7);
  int f = (int)(e & 127);
  int b = r / NPB;
  float4 xv = *(const float4*)(x + e);
  float4 hv = *(const float4*)(x + (long)b * NPB * FDIM + f);
  ushort4 o;
  o.x = bfabs_bits(hv.x - xv.x);
  o.y = bfabs_bits(hv.y - xv.y);
  o.z = bfabs_bits(hv.z - xv.z);
  o.w = bfabs_bits(hv.w - xv.w);
  *(ushort4*)(A0 + e) = o;
}

// ---------------------------------------------------------------------------
// WMMA bf16 GEMM: Y(N x COUT) = A(N x K) @ W^T, W row-major (COUT x K), bf16.
// Fused per-channel sum / sum-of-squares for BatchNorm (bias omitted: cancels
// under BN). Block = 128 threads = 4 waves, each wave one 16-row strip.
// ---------------------------------------------------------------------------
template<int K, int COUT>
__global__ __launch_bounds__(128) void k_gemm(const bf16* __restrict__ A,
                                              const bf16* __restrict__ Wb,
                                              bf16* __restrict__ Y,
                                              float* __restrict__ gsum,
                                              float* __restrict__ gsq) {
  __shared__ float s_sum[COUT];
  __shared__ float s_sq[COUT];
  const int tid = threadIdx.x;
  if (tid < COUT) { s_sum[tid] = 0.f; s_sq[tid] = 0.f; }
  __syncthreads();

  const int lane = tid & 31, wv = tid >> 5;
  const int m = lane & 15, hi = lane >> 4;
  const long row0 = (long)blockIdx.x * 64 + wv * 16;

  // A fragments: ISA 16-bit A 16x32 layout: lane half selects K {hi*8..} and {16+hi*8..}
  const bf16* arow = A + (row0 + m) * K;
  v16bf afrag[K / 32];
#pragma unroll
  for (int kt = 0; kt < K / 32; ++kt) {
    v8bf lo = *(const v8bf*)(arow + kt * 32 + hi * 8);
    v8bf hv = *(const v8bf*)(arow + kt * 32 + 16 + hi * 8);
    v16bf a;
#pragma unroll
    for (int i = 0; i < 8; ++i) { a[i] = lo[i]; a[i + 8] = hv[i]; }
    afrag[kt] = a;
  }

  v8f acc[COUT / 16];
#pragma unroll
  for (int ct = 0; ct < COUT / 16; ++ct)
#pragma unroll
    for (int i = 0; i < 8; ++i) acc[ct][i] = 0.f;

  // B fragments: 32x16 bf16, lane half selects K {hi*8..} and {16+hi*8..} (matches A)
#pragma unroll
  for (int ct = 0; ct < COUT / 16; ++ct) {
    const bf16* wrow = Wb + (ct * 16 + m) * K + hi * 8;
#pragma unroll
    for (int kt = 0; kt < K / 32; ++kt) {
      v8bf blo = *(const v8bf*)(wrow + kt * 32);
      v8bf bhi = *(const v8bf*)(wrow + kt * 32 + 16);
      v16bf bfr;
#pragma unroll
      for (int i = 0; i < 8; ++i) { bfr[i] = blo[i]; bfr[i + 8] = bhi[i]; }
      acc[ct] = __builtin_amdgcn_wmma_f32_16x16x32_bf16(
          false, afrag[kt], false, bfr, (short)0, acc[ct], false, false);
    }
  }

  // Epilogue: store Y (bf16) + per-channel BN partial stats
#pragma unroll
  for (int ct = 0; ct < COUT / 16; ++ct) {
    const int col = ct * 16 + m;
    float ls = 0.f, lq = 0.f;
#pragma unroll
    for (int r = 0; r < 8; ++r) {
      float v = acc[ct][r];
      ls += v; lq += v * v;
      Y[(row0 + hi * 8 + r) * COUT + col] = (bf16)v;
    }
    atomicAdd(&s_sum[col], ls);
    atomicAdd(&s_sq[col], lq);
  }
  __syncthreads();
  if (tid < COUT) {
    atomicAdd(&gsum[tid], s_sum[tid]);
    atomicAdd(&gsq[tid], s_sq[tid]);
  }
}

// ---------------------------------------------------------------------------
// BN finalize: scale = g*rsqrt(var+eps), shift = be - mean*scale
// ---------------------------------------------------------------------------
__global__ void k_stats(float* __restrict__ st, const float* __restrict__ g,
                        const float* __restrict__ be, int C, float invN) {
  int c = threadIdx.x;
  if (c < C) {
    float mean = st[c] * invN;
    float var  = st[128 + c] * invN - mean * mean;
    float sc   = g[c] * rsqrtf(var + 1e-5f);
    st[256 + c] = sc;
    st[384 + c] = be[c] - mean * sc;
  }
}

// ---------------------------------------------------------------------------
// In-place normalize + LeakyReLU (bf16, vectorized x8)
// ---------------------------------------------------------------------------
__global__ __launch_bounds__(256) void k_bnact(bf16* __restrict__ Y,
                                               const float* __restrict__ st, int C) {
  long base = ((long)blockIdx.x * 256 + threadIdx.x) * 8;
  int cb = (int)(base % C);
  v8bf y = *(const v8bf*)(Y + base);
  v8bf o;
#pragma unroll
  for (int i = 0; i < 8; ++i) {
    float v = (float)y[i] * st[256 + cb + i] + st[384 + cb + i];
    o[i] = (bf16)(v > 0.f ? v : 0.01f * v);
  }
  *(v8bf*)(Y + base) = o;
}

// ---------------------------------------------------------------------------
// Hub projections: P0 = hub @ WfcL^T, P1 = hub @ WfcR^T  (512 x 128 each)
// ---------------------------------------------------------------------------
__global__ __launch_bounds__(128) void k_proj(const float* __restrict__ x,
                                              const float* __restrict__ Wfc,
                                              float* __restrict__ P0,
                                              float* __restrict__ P1) {
  int b = blockIdx.x, c = threadIdx.x;
  const float* hub = x + (long)b * NPB * FDIM;
  const float* wr  = Wfc + (long)c * 256;
  float a0 = 0.f, a1 = 0.f;
#pragma unroll 8
  for (int f = 0; f < 128; ++f) {
    float h = hub[f];
    a0 += h * wr[f];
    a1 += h * wr[128 + f];
  }
  P0[b * 128 + c] = a0;
  P1[b * 128 + c] = a1;
}

// ---------------------------------------------------------------------------
// Head: s = H4 @ Wl - w_init ; global max via ordered-int atomics
// ---------------------------------------------------------------------------
__device__ inline void atomicMaxF(float* a, float v) {
  if (v >= 0.f) atomicMax((int*)a, __float_as_int(v));
  else          atomicMin((unsigned int*)a, (unsigned int)__float_as_int(v));
}

__global__ __launch_bounds__(256) void k_wl(const bf16* __restrict__ H4,
                                            const float* __restrict__ Wl,
                                            const float* __restrict__ w0,
                                            float* __restrict__ s,
                                            float* __restrict__ scal) {
  int r = blockIdx.x * 256 + threadIdx.x;
  const bf16* hrow = H4 + (long)r * 64;
  float acc = 0.f;
#pragma unroll
  for (int kk = 0; kk < 8; ++kk) {
    v8bf hv = *(const v8bf*)(hrow + kk * 8);
#pragma unroll
    for (int i = 0; i < 8; ++i) acc += (float)hv[i] * Wl[kk * 8 + i];
  }
  float sv = acc - w0[r];
  s[r] = sv;
  __shared__ float red[256];
  red[threadIdx.x] = sv; __syncthreads();
  for (int off = 128; off > 0; off >>= 1) {
    if (threadIdx.x < off) red[threadIdx.x] = fmaxf(red[threadIdx.x], red[threadIdx.x + off]);
    __syncthreads();
  }
  if (threadIdx.x == 0) atomicMaxF(&scal[0], red[0]);
}

__global__ __launch_bounds__(256) void k_exp(const float* __restrict__ s,
                                             float* __restrict__ e,
                                             float* __restrict__ scal) {
  int r = blockIdx.x * 256 + threadIdx.x;
  float ev = __expf(s[r] - scal[0]);
  e[r] = ev;
  __shared__ float red[256];
  red[threadIdx.x] = ev; __syncthreads();
  for (int off = 128; off > 0; off >>= 1) {
    if (threadIdx.x < off) red[threadIdx.x] += red[threadIdx.x + off];
    __syncthreads();
  }
  if (threadIdx.x == 0) atomicAdd(&scal[1], red[0]);
}

// ---------------------------------------------------------------------------
// Final pre-BN: pre = w0*P0[b] + softmax*P1[b]   (rank-512 replacement of the
// (N,256)@(256,128) GEMM) + fused BN stats
// ---------------------------------------------------------------------------
__global__ __launch_bounds__(128) void k_final_accum(const float* __restrict__ w0,
                                                     const float* __restrict__ e,
                                                     const float* __restrict__ scal,
                                                     const float* __restrict__ P0,
                                                     const float* __restrict__ P1,
                                                     float* __restrict__ pre,
                                                     float* __restrict__ gsum,
                                                     float* __restrict__ gsq) {
  const int c = threadIdx.x;
  const float inv = 1.f / scal[1];
  float ls = 0.f, lq = 0.f;
  const int r0 = blockIdx.x * 64;
  for (int ri = 0; ri < 64; ++ri) {
    int r = r0 + ri;
    int b = r / NPB;
    float v = w0[r] * P0[b * 128 + c] + (e[r] * inv) * P1[b * 128 + c];
    pre[(long)r * 128 + c] = v;
    ls += v; lq += v * v;
  }
  atomicAdd(&gsum[c], ls);
  atomicAdd(&gsq[c], lq);
}

__global__ __launch_bounds__(256) void k_final_bn(const float* __restrict__ pre,
                                                  const float* __restrict__ st,
                                                  float* __restrict__ out) {
  long e = ((long)blockIdx.x * 256 + threadIdx.x) * 4;
  int c = (int)(e & 127);
  float4 p = *(const float4*)(pre + e);
  float4 o;
  o.x = p.x * st[256 + c]     + st[384 + c];
  o.y = p.y * st[256 + c + 1] + st[384 + c + 1];
  o.z = p.z * st[256 + c + 2] + st[384 + c + 2];
  o.w = p.w * st[256 + c + 3] + st[384 + c + 3];
  *(float4*)(out + e) = o;
}

// ---------------------------------------------------------------------------
extern "C" void kernel_launch(void* const* d_in, const int* in_sizes, int n_in,
                              void* d_out, int out_size, void* d_ws, size_t ws_size,
                              hipStream_t stream) {
  const float* x    = (const float*)d_in[0];
  const float* wini = (const float*)d_in[1];
  const float* W1   = (const float*)d_in[2];
  const float* g1   = (const float*)d_in[4];
  const float* be1  = (const float*)d_in[5];
  const float* W2   = (const float*)d_in[6];
  const float* g2   = (const float*)d_in[8];
  const float* be2  = (const float*)d_in[9];
  const float* W3   = (const float*)d_in[10];
  const float* g3   = (const float*)d_in[12];
  const float* be3  = (const float*)d_in[13];
  const float* W4   = (const float*)d_in[14];
  const float* g4   = (const float*)d_in[16];
  const float* be4  = (const float*)d_in[17];
  const float* Wl   = (const float*)d_in[18];
  const float* Wfc  = (const float*)d_in[20];
  const float* gfc  = (const float*)d_in[22];
  const float* befc = (const float*)d_in[23];
  float* out = (float*)d_out;

  char* ws = (char*)d_ws;
  const size_t NB2 = (size_t)N_ROWS * 128 * 2;
  bf16*  buf0  = (bf16*)ws;  ws += NB2;
  bf16*  buf1  = (bf16*)ws;  ws += NB2;
  float* pre   = (float*)ws; ws += (size_t)N_ROWS * 128 * 4;
  float* P0    = (float*)ws; ws += (size_t)512 * 128 * 4;
  float* P1    = (float*)ws; ws += (size_t)512 * 128 * 4;
  bf16*  wbf   = (bf16*)ws;  ws += (size_t)45056 * 2;
  float* stats = (float*)ws; ws += (size_t)2560 * 4;   // 5 stages x 512
  float* svec  = (float*)ws; ws += (size_t)N_ROWS * 4;
  float* evec  = (float*)ws; ws += (size_t)N_ROWS * 4;
  float* scal  = (float*)ws;                            // [smax, ssum]

  const float invN = 1.f / (float)N_ROWS;
  float* st0 = stats + 0 * 512;
  float* st1 = stats + 1 * 512;
  float* st2 = stats + 2 * 512;
  float* st3 = stats + 3 * 512;
  float* st4 = stats + 4 * 512;

  k_init<<<176, 256, 0, stream>>>(W1, W2, W3, W4, wbf, stats, scal);
  k_diff<<<43264, 256, 0, stream>>>(x, (u16*)buf0);
  k_proj<<<512, 128, 0, stream>>>(x, Wfc, P0, P1);

  // Layer 1: 128 -> 128
  k_gemm<128, 128><<<5408, 128, 0, stream>>>(buf0, wbf, buf1, st0, st0 + 128);
  k_stats<<<1, 128, 0, stream>>>(st0, g1, be1, 128, invN);
  k_bnact<<<21632, 256, 0, stream>>>(buf1, st0, 128);

  // Layer 2: 128 -> 128
  k_gemm<128, 128><<<5408, 128, 0, stream>>>(buf1, wbf + 16384, buf0, st1, st1 + 128);
  k_stats<<<1, 128, 0, stream>>>(st1, g2, be2, 128, invN);
  k_bnact<<<21632, 256, 0, stream>>>(buf0, st1, 128);

  // Layer 3: 128 -> 64
  k_gemm<128, 64><<<5408, 128, 0, stream>>>(buf0, wbf + 32768, buf1, st2, st2 + 128);
  k_stats<<<1, 128, 0, stream>>>(st2, g3, be3, 64, invN);
  k_bnact<<<10816, 256, 0, stream>>>(buf1, st2, 64);

  // Layer 4: 64 -> 64
  k_gemm<64, 64><<<5408, 128, 0, stream>>>(buf1, wbf + 40960, buf0, st3, st3 + 128);
  k_stats<<<1, 128, 0, stream>>>(st3, g4, be4, 64, invN);
  k_bnact<<<10816, 256, 0, stream>>>(buf0, st3, 64);

  // Head + global softmax (2-pass)
  k_wl<<<1352, 256, 0, stream>>>(buf0, Wl, wini, svec, scal);
  k_exp<<<1352, 256, 0, stream>>>(svec, evec, scal);

  // Final rank-512 combine + BN
  k_final_accum<<<5408, 128, 0, stream>>>(wini, evec, scal, P0, P1, pre,
                                          st4, st4 + 128);
  k_stats<<<1, 128, 0, stream>>>(st4, gfc, befc, 128, invN);
  k_final_bn<<<43264, 256, 0, stream>>>(pre, st4, out);
}
